// NTXentLoss_25194278158619
// MI455X (gfx1250) — compile-verified, matrix operand order checked by reading, not run on hardware
//
#include <hip/hip_runtime.h>
#include <hip/hip_bf16.h>
#include <math.h>

// ---------------------------------------------------------------------------
// NT-Xent (SimCLR) loss, fused for MI455X (gfx1250, wave32, WMMA).
//   N = 4096, Z = 128, T = 0.5, rows = 2N = 8192.
// sim = nreps @ nreps^T never materialized. Kernel 2: each wave owns a
// 32-row A strip; the 8 waves of a block share B column-tiles staged in LDS
// (double buffered, async-to-LDS: global_load_async_to_lds_b128 + ASYNCcnt).
// Per 4KB B tile a wave does 8x v_wmma_f32_16x16x32_f16 -> 8x less L2
// traffic than private streaming. Deterministic: every scratch slot written
// exactly once; fixed-order final reduction.
// ---------------------------------------------------------------------------

typedef __attribute__((ext_vector_type(16))) _Float16 v16h;
typedef __attribute__((ext_vector_type(8)))  _Float16 h8;
typedef __attribute__((ext_vector_type(4)))  _Float16 h4;
typedef __attribute__((ext_vector_type(8)))  float    v8f;
typedef __attribute__((ext_vector_type(4)))  float    f4;
// exact pointee type the async builtin wants (per hipcc diagnostic)
typedef int v4i32 __attribute__((__vector_size__(4 * sizeof(int))));

#define NROWS   8192        // 2N
#define HALF_N  4096        // N
#define ZDIM    128
#define NCHUNK  16          // column chunks (parallelism + partial slots)
#define COLS_PER_CHUNK  (NROWS / NCHUNK)        // 512
#define TILES_PER_CHUNK (COLS_PER_CHUNK / 16)   // 32
#define WAVES_PER_BLOCK 8
#define ROWS_PER_WAVE   32                      // two 16-row M tiles
#define ROWS_PER_BLOCK  (WAVES_PER_BLOCK * ROWS_PER_WAVE)  // 256
#define LDS_PITCH 136       // 128 + 8 halves pad -> conflict-free ds_load_b128
// 1 / (T * ln2), T = 0.5  ->  exp(sim/T) == exp2(sim * SCALE)
#define EXP2_SCALE 2.8853900817779268f
#define LN2        0.6931471805599453f

// Async global->LDS staging (CDNA5, ASYNCcnt-tracked) if the toolchain has it.
#if defined(__has_builtin)
#if __has_builtin(__builtin_amdgcn_global_load_async_to_lds_b128) && \
    __has_builtin(__builtin_amdgcn_s_wait_asynccnt)
#define USE_ASYNC_LDS 1
#endif
#endif
#ifndef USE_ASYNC_LDS
#define USE_ASYNC_LDS 0
#endif

#if USE_ASYNC_LDS
#define ASYNC_G(p) ((__attribute__((address_space(1))) v4i32*)(p))
#define ASYNC_L(p) ((__attribute__((address_space(3))) v4i32*)(p))
#endif

// ---- kernel 1: row-normalize reps = concat(z1, z2); emit f32 + f16 copies --
__global__ void __launch_bounds__(256) ntxent_normalize(
    const float* __restrict__ z1, const float* __restrict__ z2,
    float* __restrict__ nf32, _Float16* __restrict__ nf16) {
  const int wave = (blockIdx.x * blockDim.x + threadIdx.x) >> 5;  // row id
  const int lane = threadIdx.x & 31;
  if (wave >= NROWS) return;
  const float* src = (wave < HALF_N) ? (z1 + (size_t)wave * ZDIM)
                                     : (z2 + (size_t)(wave - HALF_N) * ZDIM);
  f4 v = *(const f4*)(src + lane * 4);                 // 128 = 32 lanes * 4
  float ss = v.x * v.x + v.y * v.y + v.z * v.z + v.w * v.w;
  #pragma unroll
  for (int m = 16; m >= 1; m >>= 1) ss += __shfl_xor(ss, m, 32);
  const float inv = 1.0f / fmaxf(sqrtf(ss), 1e-8f);    // eps clamp as in ref
  f4 o; o.x = v.x * inv; o.y = v.y * inv; o.z = v.z * inv; o.w = v.w * inv;
  *(f4*)(nf32 + (size_t)wave * ZDIM + lane * 4) = o;
  h4 o16; o16.x = (_Float16)o.x; o16.y = (_Float16)o.y;
  o16.z = (_Float16)o.z; o16.w = (_Float16)o.w;
  *(h4*)(nf16 + (size_t)wave * ZDIM + lane * 4) = o16;
}

// ---- kernel 2: WMMA with LDS-shared, double-buffered B tiles ---------------
// Block b: strip-group sg = b / NCHUNK, chunk = b % NCHUNK.
// Wave w in block: rows [rowbase, rowbase+32), rowbase = (sg*8 + w)*32.
// A layout (16-bit A 16x32, ISA 7.12.2): lane%16 = row M; lanes 0-15 hold
// K = k0+0..7 / k0+16..23, lanes 16-31 hold K = k0+8..15 / k0+24..31.
// B (32x16) mirrors this with lane%16 = column = a row of nreps.
__global__ void __launch_bounds__(256) ntxent_simexp(
    const _Float16* __restrict__ nf16, float* __restrict__ partial) {
  __shared__ _Float16 ldsB[2][16 * LDS_PITCH];        // 2 x 4.25 KB

  const int lane   = threadIdx.x & 31;
  const int lane16 = lane & 15;
  const int half   = lane >> 4;
  const int waveid = threadIdx.x >> 5;
  const int sg     = blockIdx.x / NCHUNK;
  const int chunk  = blockIdx.x % NCHUNK;
  const int rowbase = (sg * WAVES_PER_BLOCK + waveid) * ROWS_PER_WAVE;
  const int col0    = chunk * COLS_PER_CHUNK;

  // Load the two 16x128 f16 A strips once (4 K-chunks of 32 each).
  v16h A0[4], A1[4];
  {
    const _Float16* ar0 = nf16 + (size_t)(rowbase + lane16) * ZDIM;
    const _Float16* ar1 = nf16 + (size_t)(rowbase + 16 + lane16) * ZDIM;
    #pragma unroll
    for (int c = 0; c < 4; ++c) {
      h8 lo0 = *(const h8*)(ar0 + 32 * c + 8 * half);
      h8 hi0 = *(const h8*)(ar0 + 32 * c + 8 * half + 16);
      A0[c] = __builtin_shufflevector(lo0, hi0, 0, 1, 2, 3, 4, 5, 6, 7,
                                      8, 9, 10, 11, 12, 13, 14, 15);
      h8 lo1 = *(const h8*)(ar1 + 32 * c + 8 * half);
      h8 hi1 = *(const h8*)(ar1 + 32 * c + 8 * half + 16);
      A1[c] = __builtin_shufflevector(lo1, hi1, 0, 1, 2, 3, 4, 5, 6, 7,
                                      8, 9, 10, 11, 12, 13, 14, 15);
    }
  }

  // Cooperative B staging: 256 threads x 16B == one 16x128 f16 tile (4 KB).
  const int tr = threadIdx.x >> 4;                    // tile row (column idx)
  const int ts = threadIdx.x & 15;                    // 16B segment
  const _Float16* gsrc0 = nf16 + (size_t)(col0 + tr) * ZDIM + ts * 8;
  _Float16* ldst0 = &ldsB[0][tr * LDS_PITCH + ts * 8];
  _Float16* ldst1 = &ldsB[1][tr * LDS_PITCH + ts * 8];

#if USE_ASYNC_LDS
  __builtin_amdgcn_global_load_async_to_lds_b128(ASYNC_G(gsrc0),
                                                 ASYNC_L(ldst0), 0, 0);
  __builtin_amdgcn_s_wait_asynccnt(0);
  __syncthreads();
#else
  {
    h8 r = *(const h8*)gsrc0;
    *(h8*)ldst0 = r;
  }
  __syncthreads();
#endif

  float acc0[8] = {0.f, 0.f, 0.f, 0.f, 0.f, 0.f, 0.f, 0.f};
  float acc1[8] = {0.f, 0.f, 0.f, 0.f, 0.f, 0.f, 0.f, 0.f};
  // C/D layout: slot v, this lane -> row base + v + 8*half, col lane16.
  const int diagSlot = lane16 - 8 * half;

  for (int t = 0; t < TILES_PER_CHUNK; ++t) {
    _Float16* ldst_nxt = ((t + 1) & 1) ? ldst1 : ldst0;
    const _Float16* gsrc_nxt = gsrc0 + (size_t)(t + 1) * 16 * ZDIM;
#if USE_ASYNC_LDS
    if (t + 1 < TILES_PER_CHUNK)
      __builtin_amdgcn_global_load_async_to_lds_b128(ASYNC_G(gsrc_nxt),
                                                     ASYNC_L(ldst_nxt), 0, 0);
#else
    h8 creg;
    if (t + 1 < TILES_PER_CHUNK) creg = *(const h8*)gsrc_nxt;
#endif

    // Gather all 4 B K-chunks first so ds loads clause + overlap, then burst
    // 8 WMMAs over two independent accumulate chains (c0 rows, c1 rows).
    const _Float16* lb = &ldsB[t & 1][0];
    v16h Bv[4];
    #pragma unroll
    for (int kc = 0; kc < 4; ++kc) {
      h8 lo = *(const h8*)(lb + lane16 * LDS_PITCH + 32 * kc + 8 * half);
      h8 hi = *(const h8*)(lb + lane16 * LDS_PITCH + 32 * kc + 8 * half + 16);
      Bv[kc] = __builtin_shufflevector(lo, hi, 0, 1, 2, 3, 4, 5, 6, 7,
                                       8, 9, 10, 11, 12, 13, 14, 15);
    }
    v8f c0 = {}, c1 = {};
    #pragma unroll
    for (int kc = 0; kc < 4; ++kc) {
      c0 = __builtin_amdgcn_wmma_f32_16x16x32_f16(
               false, A0[kc], false, Bv[kc], (short)0, c0, false, false);
      c1 = __builtin_amdgcn_wmma_f32_16x16x32_f16(
               false, A1[kc], false, Bv[kc], (short)0, c1, false, false);
    }

    const int colbase = col0 + t * 16;
    const bool d0 = (colbase == rowbase);
    const bool d1 = (colbase == rowbase + 16);
    #pragma unroll
    for (int v = 0; v < 8; ++v) {
      // sim in [-1,1] -> no running max needed; native v_exp_f32 (base 2)
      float e0 = __builtin_amdgcn_exp2f(c0[v] * EXP2_SCALE);
      float e1 = __builtin_amdgcn_exp2f(c1[v] * EXP2_SCALE);
      acc0[v] += (d0 && (v == diagSlot)) ? 0.0f : e0;
      acc1[v] += (d1 && (v == diagSlot)) ? 0.0f : e1;
    }

#if USE_ASYNC_LDS
    __builtin_amdgcn_s_wait_asynccnt(0);   // staged writes landed
    __syncthreads();                       // ...and visible to all waves
#else
    __syncthreads();                       // readers done with next buffer
    if (t + 1 < TILES_PER_CHUNK) *(h8*)ldst_nxt = creg;
    __syncthreads();                       // stores visible
#endif
  }

  // Row sums live across the 16 lanes of a half-wave (fixed slot v).
  #pragma unroll
  for (int v = 0; v < 8; ++v) {
    float s0 = acc0[v], s1 = acc1[v];
    #pragma unroll
    for (int m = 8; m >= 1; m >>= 1) {
      s0 += __shfl_xor(s0, m, 32);
      s1 += __shfl_xor(s1, m, 32);
    }
    if (lane16 == 0) {
      partial[(size_t)(rowbase + v + 8 * half) * NCHUNK + chunk] = s0;
      partial[(size_t)(rowbase + 16 + v + 8 * half) * NCHUNK + chunk] = s1;
    }
  }
}

// ---- kernel 3: per-row finalize: lse - pos/T -------------------------------
__global__ void __launch_bounds__(256) ntxent_rows(
    const float* __restrict__ nf32, const float* __restrict__ partial,
    float* __restrict__ perrow) {
  const int wave = (blockIdx.x * blockDim.x + threadIdx.x) >> 5;  // row
  const int lane = threadIdx.x & 31;
  if (wave >= NROWS) return;
  const int partner = (wave < HALF_N) ? wave + HALF_N : wave - HALF_N;
  f4 a = *(const f4*)(nf32 + (size_t)wave * ZDIM + lane * 4);
  f4 b = *(const f4*)(nf32 + (size_t)partner * ZDIM + lane * 4);
  float d  = a.x * b.x + a.y * b.y + a.z * b.z + a.w * b.w;   // positive (f32)
  float ps = (lane < NCHUNK) ? partial[(size_t)wave * NCHUNK + lane] : 0.0f;
  #pragma unroll
  for (int m = 16; m >= 1; m >>= 1) {
    d  += __shfl_xor(d, m, 32);
    ps += __shfl_xor(ps, m, 32);
  }
  if (lane == 0) {
    // ps = sum exp2(sim*SCALE) -> ln(sum exp(sim/T)) = log2(ps) * ln2
    float lse = __builtin_amdgcn_logf(ps) * LN2;
    perrow[wave] = lse - d * 2.0f;                  // pos / T, T = 0.5
  }
}

// ---- kernel 4: fixed-order scalar reduction -> mean over 2N ---------------
__global__ void __launch_bounds__(256) ntxent_reduce(
    const float* __restrict__ perrow, float* __restrict__ out) {
  __shared__ float red[256];
  float s = 0.0f;
  for (int i = threadIdx.x; i < NROWS; i += 256) s += perrow[i];
  red[threadIdx.x] = s;
  __syncthreads();
  #pragma unroll
  for (int step = 128; step > 0; step >>= 1) {
    if ((int)threadIdx.x < step) red[threadIdx.x] += red[threadIdx.x + step];
    __syncthreads();
  }
  if (threadIdx.x == 0) out[0] = red[0] * (1.0f / (float)NROWS);
}

extern "C" void kernel_launch(void* const* d_in, const int* in_sizes, int n_in,
                              void* d_out, int out_size, void* d_ws,
                              size_t ws_size, hipStream_t stream) {
  const float* z1 = (const float*)d_in[0];
  const float* z2 = (const float*)d_in[1];
  float* out = (float*)d_out;

  // workspace carve-up (all 256B aligned):
  //   nf32    : 8192*128*4 = 4 MiB      @ 0
  //   nf16    : 8192*128*2 = 2 MiB      @ 4 MiB
  //   partial : 8192*16*4  = 512 KiB    @ 6 MiB
  //   perrow  : 8192*4     = 32 KiB     @ 6.5 MiB
  char* ws = (char*)d_ws;
  float*    nf32    = (float*)ws;
  _Float16* nf16    = (_Float16*)(ws + (4u << 20));
  float*    partial = (float*)(ws + (6u << 20));
  float*    perrow  = (float*)(ws + (6u << 20) + (512u << 10));

  // 8192 row-waves, 8 waves per 256-thread block
  ntxent_normalize<<<NROWS / 8, 256, 0, stream>>>(z1, z2, nf32, nf16);
  // 32 strip-groups x 16 chunks = 512 blocks, 8 waves each
  ntxent_simexp<<<(NROWS / ROWS_PER_BLOCK) * NCHUNK, 256, 0, stream>>>(
      nf16, partial);
  ntxent_rows<<<NROWS / 8, 256, 0, stream>>>(nf32, partial, perrow);
  ntxent_reduce<<<1, 256, 0, stream>>>(perrow, out);
}
